// ContextualGraphFusion_67534065762581
// MI455X (gfx1250) — matrix operand order repaired and checked by reading, not compiled, not run on hardware
//
#include <hip/hip_runtime.h>

// ---------------- problem constants ----------------
#define Bd   32
#define Nd   512
#define FDd  768
#define HDd  384
#define Gd   3
#define Md   (Bd * Nd)          // 16384 rows
#define ALPHA 0.2f
#define NEGV  (-9.0e15f)

typedef __attribute__((ext_vector_type(16))) _Float16 v16h;
typedef __attribute__((ext_vector_type(8)))  float    v8f;

// ---------------- wave reductions (wave32) ----------------
__device__ __forceinline__ float wredSum(float v) {
#pragma unroll
  for (int i = 16; i > 0; i >>= 1) v += __shfl_xor(v, i, 32);
  return v;
}
__device__ __forceinline__ float wredMax(float v) {
#pragma unroll
  for (int i = 16; i > 0; i >>= 1) v = fmaxf(v, __shfl_xor(v, i, 32));
  return v;
}

// ---------------- WMMA helpers ----------------
__device__ __forceinline__ v8f wmma_f16(v16h a, v16h b, v8f c) {
  return __builtin_amdgcn_wmma_f32_16x16x32_f16(
      false, a, false, b, (short)0, c, false, false);
}

// A fragment: 16x32 f16 tile, row-major with `stride` halfs per row.
// lane holds row M=lane%16; lanes0-15 K in {0..7,16..23}, lanes16-31 {8..15,24..31}.
// Each half of the fragment is 16B contiguous -> 2x b128 loads (LDS or global).
__device__ __forceinline__ v16h frag_a(const _Float16* p, int stride) {
  const int lane = threadIdx.x & 31;
  union { v16h v; uint4 q[2]; } f;
  const int m  = lane & 15;
  const int kb = (lane & 16) ? 8 : 0;
  const _Float16* rp = p + m * stride + kb;
  f.q[0] = *(const uint4*)(rp);
  f.q[1] = *(const uint4*)(rp + 16);
  return f.v;
}

// B fragment from a K-contiguous (transposed) operand: p points at (col0,k0);
// column n = col0+lane%16 is a row of `stride` halfs; lanes0-15 read K in {0..7,16..23},
// lanes16-31 K in {8..15,24..31} -> 2x b128 loads.
__device__ __forceinline__ v16h frag_b(const _Float16* p, int stride) {
  const int lane = threadIdx.x & 31;
  union { v16h v; uint4 q[2]; } f;
  const int n  = lane & 15;
  const int kb = (lane & 16) ? 8 : 0;
  const _Float16* cp = p + (size_t)n * stride + kb;
  f.q[0] = *(const uint4*)(cp);
  f.q[1] = *(const uint4*)(cp + 16);
  return f.v;
}

// ---------------- prep: f32 -> f16 copy ----------------
__global__ __launch_bounds__(256) void cvt_f16_kernel(
    const float* __restrict__ in, _Float16* __restrict__ out, int n) {
  for (int i = blockIdx.x * 256 + threadIdx.x; i < n; i += gridDim.x * 256)
    out[i] = (_Float16)in[i];
}

// prep: transpose-convert weights: out[z][n][k] = (f16) in[z][k][n]
__global__ __launch_bounds__(256) void cvtT_kernel(
    const float* __restrict__ in, _Float16* __restrict__ out, int K, int Nc) {
  const float* src = in + (size_t)blockIdx.z * K * Nc;
  _Float16* dst    = out + (size_t)blockIdx.z * K * Nc;
  const int n = blockIdx.x;
  for (int k = threadIdx.x; k < K; k += 256)
    dst[(size_t)n * K + k] = (_Float16)src[(size_t)k * Nc + n];
}

// ---------------- K1: uncertainty weights ----------------
__global__ void attw_kernel(const float* __restrict__ log_unc,
                            float* __restrict__ attw) {
  if (threadIdx.x == 0) {
    float s[Gd];
#pragma unroll
    for (int g = 0; g < Gd; ++g) s[g] = 1.0f / (__expf(log_unc[g]) + 1e-8f);
    float m = fmaxf(s[0], fmaxf(s[1], s[2]));
    float e0 = __expf(s[0] - m), e1 = __expf(s[1] - m), e2 = __expf(s[2] - m);
    float inv = 1.0f / (e0 + e1 + e2);
    attw[0] = e0 * inv; attw[1] = e1 * inv; attw[2] = e2 * inv;
  }
}

// ---------------- K2: content gates cg[n,g] ----------------
__global__ __launch_bounds__(128) void gate_kernel(
    const float* __restrict__ x, const float* __restrict__ gate_w,
    const float* __restrict__ gate_b, float* __restrict__ cg) {
  __shared__ float red[Gd][128];
  const int n = blockIdx.x, tid = threadIdx.x;
  float a0 = 0.f, a1 = 0.f, a2 = 0.f;
  for (int f = tid; f < FDd; f += 128) {
    float av = 0.f;
#pragma unroll
    for (int b = 0; b < Bd; ++b) av += x[((size_t)b * Nd + n) * FDd + f];
    av *= (1.0f / Bd);
    a0 += av * gate_w[f * Gd + 0];
    a1 += av * gate_w[f * Gd + 1];
    a2 += av * gate_w[f * Gd + 2];
  }
  red[0][tid] = a0; red[1][tid] = a1; red[2][tid] = a2;
  __syncthreads();
  for (int s = 64; s > 0; s >>= 1) {
    if (tid < s) {
      red[0][tid] += red[0][tid + s];
      red[1][tid] += red[1][tid + s];
      red[2][tid] += red[2][tid + s];
    }
    __syncthreads();
  }
  if (tid == 0) {
    float l0 = red[0][0] + gate_b[0];
    float l1 = red[1][0] + gate_b[1];
    float l2 = red[2][0] + gate_b[2];
    float m  = fmaxf(l0, fmaxf(l1, l2));
    float e0 = __expf(l0 - m), e1 = __expf(l1 - m), e2 = __expf(l2 - m);
    float inv = 1.0f / (e0 + e1 + e2);
    cg[n * Gd + 0] = e0 * inv;
    cg[n * Gd + 1] = e1 * inv;
    cg[n * Gd + 2] = e2 * inv;
  }
}

// ---------------- K3: h^T[g,b,d,n] = (x @ W[g])^T, f16 -------------------
// All operands f16 & L2-resident; fragments straight from global (2x b128
// each), zero LDS staging, zero k-loop barriers. LDS used only to transpose
// the 128x64 output tile so hT stores are 256B-contiguous along n.
__global__ __launch_bounds__(256) void gemm_h_kernel(
    const _Float16* __restrict__ xh, const _Float16* __restrict__ WT,
    _Float16* __restrict__ hT) {
  __shared__ __align__(16) _Float16 Ct[64 * 128];          // 16 KB
  const int tid = threadIdx.x;
  const int g  = blockIdx.z;
  const int rm = blockIdx.x * 128;
  const int cn = blockIdx.y * 64;
  const int wave = tid >> 5, lane = tid & 31;
  const int wm = wave & 3, wn = wave >> 2;
  const _Float16* ap = xh + (size_t)(rm + wm * 32) * FDd;
  const _Float16* bp = WT + (size_t)g * FDd * HDd + (size_t)(cn + wn * 32) * FDd;
  v8f acc[2][2] = {};
  for (int kc = 0; kc < FDd / 32; ++kc) {
    const int k0 = kc * 32;
    v16h af0 = frag_a(ap + k0, FDd);
    v16h af1 = frag_a(ap + (size_t)16 * FDd + k0, FDd);
    v16h bf0 = frag_b(bp + k0, FDd);
    v16h bf1 = frag_b(bp + (size_t)16 * FDd + k0, FDd);
    acc[0][0] = wmma_f16(af0, bf0, acc[0][0]);
    acc[0][1] = wmma_f16(af0, bf1, acc[0][1]);
    acc[1][0] = wmma_f16(af1, bf0, acc[1][0]);
    acc[1][1] = wmma_f16(af1, bf1, acc[1][1]);
  }
  // transpose 128x64 tile through LDS
  const int mo = (lane & 16) ? 8 : 0;
#pragma unroll
  for (int i = 0; i < 2; ++i)
#pragma unroll
    for (int j = 0; j < 2; ++j)
#pragma unroll
      for (int r = 0; r < 8; ++r) {
        int ml = wm * 32 + i * 16 + r + mo;
        int cl = wn * 32 + j * 16 + (lane & 15);
        Ct[cl * 128 + ml] = (_Float16)acc[i][j][r];
      }
  __syncthreads();
  const int b  = rm / Nd;
  const int i0 = rm % Nd;
  const uint4* srcq = (const uint4*)Ct;
  for (int e = tid; e < 1024; e += 256) {       // 8192 halfs = 1024 uint4
    int c = e >> 4, rq = e & 15;
    _Float16* dst = hT + ((size_t)(g * Bd + b) * HDd + cn + c) * Nd + i0 + rq * 8;
    *(uint4*)dst = srcq[e];
  }
}

// ---------------- K4: e1/e2 from transposed h ----------------
__global__ __launch_bounds__(256) void e12_kernel(
    const _Float16* __restrict__ hT, const float* __restrict__ a,
    float* __restrict__ e1, float* __restrict__ e2) {
  const int gb = blockIdx.x;                 // g*B + b
  const int g  = gb >> 5;
  const _Float16* base = hT + (size_t)gb * HDd * Nd;
  const float* ag = a + g * (2 * HDd);
  const int n0 = threadIdx.x, n1 = threadIdx.x + 256;
  float s1a = 0.f, s2a = 0.f, s1b = 0.f, s2b = 0.f;
  for (int d = 0; d < HDd; ++d) {
    const _Float16* row = base + (size_t)d * Nd;
    float h0 = (float)row[n0], h1 = (float)row[n1];
    float a1v = ag[d], a2v = ag[HDd + d];
    s1a += h0 * a1v; s2a += h0 * a2v;
    s1b += h1 * a1v; s2b += h1 * a2v;
  }
  const size_t eb = (size_t)gb * Nd;
  e1[eb + n0] = s1a; e2[eb + n0] = s2a;
  e1[eb + n1] = s1b; e2[eb + n1] = s2b;
}

// ---------------- K5: fused masked-softmax attention + ELU + LN + mix -----
__global__ __launch_bounds__(256) void attn_kernel(
    const _Float16* __restrict__ hT, const float* __restrict__ e1,
    const float* __restrict__ e2, const int* __restrict__ adj,
    const float* __restrict__ attw, const float* __restrict__ cg,
    const float* __restrict__ ln1g, const float* __restrict__ ln1b,
    _Float16* __restrict__ fused) {
  __shared__ __align__(16) _Float16 attS[16 * Nd];   // 16 KB
  __shared__ float outS[16 * HDd];                   // 24 KB
  const int tid = threadIdx.x, wave = tid >> 5, lane = tid & 31;
  const int it = blockIdx.x, b = blockIdx.y;
  float facc[2][12] = {};

  for (int g = 0; g < Gd; ++g) {
    const size_t ebase = (size_t)(g * Bd + b) * Nd;
    float e2v[16];
#pragma unroll
    for (int k = 0; k < 16; ++k) e2v[k] = e2[ebase + lane + 32 * k];

    // masked leaky-relu scores + row softmax (wave owns rows wave, wave+8)
#pragma unroll
    for (int ri = 0; ri < 2; ++ri) {
      const int rr = wave + ri * 8;
      const int i = it * 16 + rr;
      const float e1v = e1[ebase + i];
      const int* adjrow = adj + (size_t)g * Nd * Nd + (size_t)i * Nd;
      float ev[16];
      float mx = -3.0e38f;
#pragma unroll
      for (int k = 0; k < 16; ++k) {
        int j = lane + 32 * k;
        float s = e1v + e2v[k];
        s = (s > 0.f) ? s : ALPHA * s;
        if (adjrow[j] <= 0) s = NEGV;
        ev[k] = s;
        mx = fmaxf(mx, s);
      }
      mx = wredMax(mx);
      float sum = 0.f;
#pragma unroll
      for (int k = 0; k < 16; ++k) { ev[k] = __expf(ev[k] - mx); sum += ev[k]; }
      sum = wredSum(sum);
      const float inv = 1.0f / sum;
#pragma unroll
      for (int k = 0; k < 16; ++k)
        attS[rr * Nd + lane + 32 * k] = (_Float16)(ev[k] * inv);
    }
    __syncthreads();

    // h' = att(16xN) @ h(N x HD): A from LDS, B straight from global hT
    v8f acc[3] = {};
    const _Float16* hTb = hT + (size_t)(g * Bd + b) * HDd * Nd;
    for (int kc = 0; kc < Nd / 32; ++kc) {
      v16h af = frag_a(attS + kc * 32, Nd);
#pragma unroll
      for (int t = 0; t < 3; ++t) {
        v16h bf = frag_b(hTb + (size_t)((wave * 3 + t) * 16) * Nd + kc * 32, Nd);
        acc[t] = wmma_f16(af, bf, acc[t]);
      }
    }
    __syncthreads();

    // ELU into LDS
    const int mo = (lane & 16) ? 8 : 0;
#pragma unroll
    for (int t = 0; t < 3; ++t)
#pragma unroll
      for (int r = 0; r < 8; ++r) {
        int m = r + mo;
        int c = (wave * 3 + t) * 16 + (lane & 15);
        float v = acc[t][r];
        outS[m * HDd + c] = (v > 0.f) ? v : (__expf(v) - 1.0f);
      }
    __syncthreads();

    // per-row LayerNorm(HD) + weighted accumulation
#pragma unroll
    for (int ri = 0; ri < 2; ++ri) {
      const int rr = wave + ri * 8;
      const float wt = 0.7f * attw[g] + 0.3f * cg[(it * 16 + rr) * Gd + g];
      float vv[12], s = 0.f, sq = 0.f;
#pragma unroll
      for (int k = 0; k < 12; ++k) {
        vv[k] = outS[rr * HDd + lane + 32 * k];
        s += vv[k]; sq += vv[k] * vv[k];
      }
      s = wredSum(s); sq = wredSum(sq);
      const float mean = s * (1.0f / HDd);
      const float var  = sq * (1.0f / HDd) - mean * mean;
      const float rinv = rsqrtf(var + 1e-5f);
#pragma unroll
      for (int k = 0; k < 12; ++k) {
        int d = lane + 32 * k;
        facc[ri][k] += wt * ((vv[k] - mean) * rinv * ln1g[d] + ln1b[d]);
      }
    }
    __syncthreads();
  }

#pragma unroll
  for (int ri = 0; ri < 2; ++ri) {
    const int rr = wave + ri * 8;
    _Float16* dst = fused + ((size_t)b * Nd + it * 16 + rr) * HDd;
#pragma unroll
    for (int k = 0; k < 12; ++k) dst[lane + 32 * k] = (_Float16)facc[ri][k];
  }
}

// ---------------- K6: t1 = relu(fused @ w1 + b1), f16 out -----------------
// Barrier-free: A (fused f16) and B (w1^T f16) fragments from global/L2.
__global__ __launch_bounds__(256) void gemm_t1_kernel(
    const _Float16* __restrict__ fused, const _Float16* __restrict__ w1T,
    const float* __restrict__ b1, _Float16* __restrict__ t1) {
  const int tid = threadIdx.x;
  const int rm = blockIdx.x * 128;
  const int cn = blockIdx.y * 64;
  const int wave = tid >> 5, lane = tid & 31;
  const int wm = wave & 3, wn = wave >> 2;
  const _Float16* ap = fused + (size_t)(rm + wm * 32) * HDd;
  const _Float16* bp = w1T + (size_t)(cn + wn * 32) * HDd;
  v8f acc[2][2] = {};
  for (int kc = 0; kc < HDd / 32; ++kc) {
    const int k0 = kc * 32;
    v16h af0 = frag_a(ap + k0, HDd);
    v16h af1 = frag_a(ap + (size_t)16 * HDd + k0, HDd);
    v16h bf0 = frag_b(bp + k0, HDd);
    v16h bf1 = frag_b(bp + (size_t)16 * HDd + k0, HDd);
    acc[0][0] = wmma_f16(af0, bf0, acc[0][0]);
    acc[0][1] = wmma_f16(af0, bf1, acc[0][1]);
    acc[1][0] = wmma_f16(af1, bf0, acc[1][0]);
    acc[1][1] = wmma_f16(af1, bf1, acc[1][1]);
  }
  const int mo = (lane & 16) ? 8 : 0;
#pragma unroll
  for (int i = 0; i < 2; ++i)
#pragma unroll
    for (int j = 0; j < 2; ++j)
#pragma unroll
      for (int r = 0; r < 8; ++r) {
        int m = rm + wm * 32 + i * 16 + r + mo;
        int c = cn + wn * 32 + j * 16 + (lane & 15);
        t1[(size_t)m * HDd + c] = (_Float16)fmaxf(acc[i][j][r] + b1[c], 0.0f);
      }
}

// ---------------- K7: proj = t1@w2 + b2 -> LN(FD) -> residual gate --------
__global__ __launch_bounds__(256) void final_kernel(
    const _Float16* __restrict__ t1, const _Float16* __restrict__ w2T,
    const float* __restrict__ b2, const float* __restrict__ ln2g,
    const float* __restrict__ ln2b, const float* __restrict__ x,
    float* __restrict__ out) {
  __shared__ float outS[16 * FDd];                          // 48 KB
  const int tid = threadIdx.x, wave = tid >> 5, lane = tid & 31;
  const int rm = blockIdx.x * 16;
  const _Float16* ap = t1 + (size_t)rm * HDd;
  v8f acc[6] = {};
  for (int kc = 0; kc < HDd / 32; ++kc) {
    const int k0 = kc * 32;
    v16h af = frag_a(ap + k0, HDd);
#pragma unroll
    for (int t = 0; t < 6; ++t) {
      v16h bf = frag_b(w2T + (size_t)((wave * 6 + t) * 16) * HDd + k0, HDd);
      acc[t] = wmma_f16(af, bf, acc[t]);
    }
  }
  const int mo = (lane & 16) ? 8 : 0;
#pragma unroll
  for (int t = 0; t < 6; ++t)
#pragma unroll
    for (int r = 0; r < 8; ++r) {
      int m = r + mo;
      int c = (wave * 6 + t) * 16 + (lane & 15);
      outS[m * FDd + c] = acc[t][r] + b2[c];
    }
  __syncthreads();

#pragma unroll
  for (int ri = 0; ri < 2; ++ri) {
    const int rr = wave + ri * 8;
    const int row = rm + rr;
    float pv[24], xv[24], s = 0.f, sq = 0.f;
#pragma unroll
    for (int k = 0; k < 24; ++k) {
      pv[k] = outS[rr * FDd + lane + 32 * k];
      s += pv[k]; sq += pv[k] * pv[k];
    }
    s = wredSum(s); sq = wredSum(sq);
    const float mean = s * (1.0f / FDd);
    const float var  = sq * (1.0f / FDd) - mean * mean;
    const float rinv = rsqrtf(var + 1e-5f);
    float dot = 0.f;
#pragma unroll
    for (int k = 0; k < 24; ++k) {
      int d = lane + 32 * k;
      pv[k] = (pv[k] - mean) * rinv * ln2g[d] + ln2b[d];
      xv[k] = x[(size_t)row * FDd + d];
      dot += pv[k] * xv[k];
    }
    dot = wredSum(dot);
    const float gv = 1.0f / (1.0f + __expf(-dot * 0.03608439182435161f)); // 1/sqrt(768)
#pragma unroll
    for (int k = 0; k < 24; ++k) {
      int d = lane + 32 * k;
      out[(size_t)row * FDd + d] = gv * xv[k] + (1.0f - gv) * pv[k];
    }
  }
}

// ---------------- host-side launch ----------------
extern "C" void kernel_launch(void* const* d_in, const int* in_sizes, int n_in,
                              void* d_out, int out_size, void* d_ws, size_t ws_size,
                              hipStream_t stream) {
  (void)in_sizes; (void)n_in; (void)out_size; (void)ws_size;
  const float* x        = (const float*)d_in[0];
  const int*   adj      = (const int*)d_in[1];
  const float* W        = (const float*)d_in[2];
  const float* a        = (const float*)d_in[3];
  const float* log_unc  = (const float*)d_in[4];
  const float* gate_w   = (const float*)d_in[5];
  const float* gate_b   = (const float*)d_in[6];
  const float* ln1_g    = (const float*)d_in[7];
  const float* ln1_b    = (const float*)d_in[8];
  const float* w1       = (const float*)d_in[9];
  const float* b1       = (const float*)d_in[10];
  const float* w2       = (const float*)d_in[11];
  const float* b2       = (const float*)d_in[12];
  const float* ln2_g    = (const float*)d_in[13];
  const float* ln2_b    = (const float*)d_in[14];
  float* out = (float*)d_out;

  // workspace carve-out (~91 MB)
  char* ws = (char*)d_ws;
  size_t o = 0;
  float* attw   = (float*)(ws + o);      o += 256;
  float* cg     = (float*)(ws + o);      o += (size_t)Nd * Gd * 4;
  float* e1     = (float*)(ws + o);      o += (size_t)Gd * Md * 4;
  float* e2     = (float*)(ws + o);      o += (size_t)Gd * Md * 4;
  _Float16* hT  = (_Float16*)(ws + o);   o += (size_t)Gd * Md * HDd * 2;  // [G,B,HD,N]
  _Float16* xh  = (_Float16*)(ws + o);   o += (size_t)Md * FDd * 2;       // x in f16
  _Float16* WT  = (_Float16*)(ws + o);   o += (size_t)Gd * FDd * HDd * 2; // [G,HD,FD]
  _Float16* w1T = (_Float16*)(ws + o);   o += (size_t)HDd * HDd * 2;      // [HD,HD]
  _Float16* w2T = (_Float16*)(ws + o);   o += (size_t)HDd * FDd * 2;      // [FD,HD]
  _Float16* fused = (_Float16*)(ws + o); o += (size_t)Md * HDd * 2;
  _Float16* t1f   = (_Float16*)(ws + o); o += (size_t)Md * HDd * 2;

  // one-time conversions / transposes (all L2-resident afterwards)
  cvt_f16_kernel<<<4096, 256, 0, stream>>>(x, xh, Md * FDd);
  cvtT_kernel<<<dim3(HDd, 1, Gd), 256, 0, stream>>>(W, WT, FDd, HDd);
  cvtT_kernel<<<dim3(HDd, 1, 1), 256, 0, stream>>>(w1, w1T, HDd, HDd);
  cvtT_kernel<<<dim3(FDd, 1, 1), 256, 0, stream>>>(w2, w2T, HDd, FDd);

  attw_kernel<<<1, 32, 0, stream>>>(log_unc, attw);
  gate_kernel<<<Nd, 128, 0, stream>>>(x, gate_w, gate_b, cg);
  gemm_h_kernel<<<dim3(Md / 128, HDd / 64, Gd), 256, 0, stream>>>(xh, WT, hT);
  e12_kernel<<<Gd * Bd, 256, 0, stream>>>(hT, a, e1, e2);
  attn_kernel<<<dim3(Nd / 16, Bd), 256, 0, stream>>>(
      hT, e1, e2, adj, attw, cg, ln1_g, ln1_b, fused);
  gemm_t1_kernel<<<dim3(Md / 128, HDd / 64), 256, 0, stream>>>(fused, w1T, b1, t1f);
  final_kernel<<<Md / 16, 256, 0, stream>>>(t1f, w2T, b2, ln2_g, ln2_b, x, out);
}
